// PNT_2_38250978738808
// MI455X (gfx1250) — matrix operand order repaired and checked by reading, not compiled, not run on hardware
//
#include <hip/hip_runtime.h>

typedef __attribute__((ext_vector_type(16))) _Float16 v16h;
typedef __attribute__((ext_vector_type(8)))  _Float16 v8h;
typedef __attribute__((ext_vector_type(8)))  float    v8f;

#define BN_EPS 1e-5f

// ---------------- transpose [B,C,N] -> [B,N,C] ----------------
__global__ void k_t_in(const float* __restrict__ in, float* __restrict__ out,
                       int B, int C, int N) {
  size_t i = (size_t)blockIdx.x * blockDim.x + threadIdx.x;
  size_t tot = (size_t)B * C * N;
  if (i >= tot) return;
  int n = (int)(i % N);
  int c = (int)((i / N) % C);
  int b = (int)(i / ((size_t)C * N));
  out[((size_t)b * N + n) * C + c] = in[i];
}

// ---------------- transpose [B,S,C] -> [B,C,S] ----------------
__global__ void k_t_out(const float* __restrict__ in, float* __restrict__ out,
                        int B, int S, int C) {
  size_t i = (size_t)blockIdx.x * blockDim.x + threadIdx.x;
  size_t tot = (size_t)B * S * C;
  if (i >= tot) return;
  int c = (int)(i % C);
  int s = (int)((i / C) % S);
  int b = (int)(i / ((size_t)S * C));
  out[((size_t)b * C + c) * S + s] = in[i];
}

__global__ void k_zero(float* __restrict__ p, int n) {
  int i = blockIdx.x * blockDim.x + threadIdx.x;
  if (i < n) p[i] = 0.f;
}

// ---------------- farthest point sampling ----------------
__device__ __forceinline__ void wave_argmax(float& d, int& i) {
#pragma unroll
  for (int off = 16; off > 0; off >>= 1) {
    float od = __shfl_down(d, off, 32);
    int   oi = __shfl_down(i, off, 32);
    if (od > d || (od == d && oi < i)) { d = od; i = oi; }
  }
}

template <int PPT>
__global__ void k_fps(const float* __restrict__ xyz, int N, int S,
                      int* __restrict__ outIdx, float* __restrict__ newXyz) {
  __shared__ float sd[32];
  __shared__ int   si[32];
  __shared__ int   sfar;
  const int b    = blockIdx.x;
  const int tid  = threadIdx.x;
  const int lane = tid & 31, wave = tid >> 5;
  const float* base = xyz + (size_t)b * N * 3;
  float px[PPT], py[PPT], pz[PPT], dist[PPT];
#pragma unroll
  for (int p = 0; p < PPT; ++p) {
    int j = tid + p * 1024;
    if (j < N) {
      px[p] = base[j * 3 + 0];
      py[p] = base[j * 3 + 1];
      pz[p] = base[j * 3 + 2];
      dist[p] = 1e10f;
    }
  }
  if (tid == 0) sfar = 0;
  __syncthreads();
  for (int s = 0; s < S; ++s) {
    int far = sfar;
    float cx = base[far * 3 + 0], cy = base[far * 3 + 1], cz = base[far * 3 + 2];
    if (tid == 0) {
      outIdx[(size_t)b * S + s] = far;
      newXyz[((size_t)b * S + s) * 3 + 0] = cx;
      newXyz[((size_t)b * S + s) * 3 + 1] = cy;
      newXyz[((size_t)b * S + s) * 3 + 2] = cz;
    }
    float bd = -1.0f; int bi = 0x7fffffff;
#pragma unroll
    for (int p = 0; p < PPT; ++p) {
      int j = tid + p * 1024;
      if (j < N) {
        float dx = px[p] - cx, dy = py[p] - cy, dz = pz[p] - cz;
        float d  = dx * dx + dy * dy + dz * dz;
        float nd = fminf(dist[p], d);
        dist[p] = nd;
        if (nd > bd || (nd == bd && j < bi)) { bd = nd; bi = j; }
      }
    }
    wave_argmax(bd, bi);
    if (lane == 0) { sd[wave] = bd; si[wave] = bi; }
    __syncthreads();
    if (wave == 0) {
      float d2 = sd[lane]; int i2 = si[lane];
      wave_argmax(d2, i2);
      if (lane == 0) sfar = i2;
    }
    __syncthreads();
  }
}

// ---------------- ball query + grouping (writes padded f16 GEMM rows) ----------------
__device__ __forceinline__ void write_row(_Float16* __restrict__ X, size_t rowOff,
                                          const float* __restrict__ pts,
                                          const float* __restrict__ xyz,
                                          int j, int D, int Cpad,
                                          float cx, float cy, float cz) {
  const float* pr = pts + (size_t)j * D;
  _Float16* dst = X + rowOff;
  for (int c = 0; c < D; ++c) dst[c] = (_Float16)pr[c];
  const float* xr = xyz + (size_t)j * 3;
  dst[D + 0] = (_Float16)(xr[0] - cx);
  dst[D + 1] = (_Float16)(xr[1] - cy);
  dst[D + 2] = (_Float16)(xr[2] - cz);
  for (int c = D + 3; c < Cpad; ++c) dst[c] = (_Float16)0.f;
}

__global__ void k_ball_group(const float* __restrict__ xyz, const float* __restrict__ nxyz,
                             const float* __restrict__ pts,
                             int N, int S, int K, int D, int Cpad, float r2,
                             _Float16* __restrict__ X) {
  const int lane = threadIdx.x & 31;
  const int w = blockIdx.x * (blockDim.x >> 5) + (threadIdx.x >> 5);
  const int b = w / S;
  const float* xb = xyz + (size_t)b * N * 3;
  const float* pb = pts + (size_t)b * N * D;
  float cx = nxyz[(size_t)w * 3 + 0];
  float cy = nxyz[(size_t)w * 3 + 1];
  float cz = nxyz[(size_t)w * 3 + 2];
  size_t rowBase = (size_t)w * K;
  int count = 0, first = -1;
  for (int base = 0; base < N; base += 32) {
    int j = base + lane;                       // N is a multiple of 32
    float dx = xb[j * 3 + 0] - cx;
    float dy = xb[j * 3 + 1] - cy;
    float dz = xb[j * 3 + 2] - cz;
    float d2 = dx * dx + dy * dy + dz * dz;
    bool in = (d2 <= r2);
    unsigned mask = (unsigned)__ballot(in);
    if (in) {
      int slot = count + __popc(mask & ((1u << lane) - 1u));
      if (slot < K)
        write_row(X, (rowBase + slot) * Cpad, pb, xb, j, D, Cpad, cx, cy, cz);
    }
    if (first < 0 && mask) first = base + __ffs(mask) - 1;
    count += __popc(mask);
    if (count >= K) break;                     // uniform across the wave
  }
  if (count < K) {
    int src = (first >= 0) ? first : 0;        // reference pads with first in-ball index
    for (int slot = count + lane; slot < K; slot += 32)
      write_row(X, (rowBase + slot) * Cpad, pb, xb, src, D, Cpad, cx, cy, cz);
  }
}

// ---------------- weight pack f32[Co,Cin] -> f16[Co,Cpad] ----------------
__global__ void k_pack_w(const float* __restrict__ Wsrc, _Float16* __restrict__ Wt,
                         int Co, int Cin, int Cpad) {
  int i = blockIdx.x * blockDim.x + threadIdx.x;
  if (i >= Co * Cpad) return;
  int ci = i % Cpad, co = i / Cpad;
  Wt[i] = (ci < Cin) ? (_Float16)Wsrc[co * Cin + ci] : (_Float16)0.f;
}

// ---------------- WMMA GEMM: Y[M,Co] = X[M,Cpad] * W[Co,Cpad]^T + bias ----------------
__global__ void k_gemm(const _Float16* __restrict__ X, const _Float16* __restrict__ Wt,
                       const float* __restrict__ bias, float* __restrict__ Y,
                       int M, int Cpad, int Co) {
  const int lane  = threadIdx.x & 31;
  const int wave  = threadIdx.x >> 5;
  const int mtile = blockIdx.x * 4 + wave;
  const int ntile = blockIdx.y;
  if (mtile * 16 >= M) return;                 // uniform per wave -> EXEC stays all-1s
  const int l15 = lane & 15;
  const int hi  = lane >> 4;
  const _Float16* xr = X  + (size_t)(mtile * 16 + l15) * Cpad;
  const _Float16* wr = Wt + (size_t)(ntile * 16 + l15) * Cpad;
  v8f c = {0.f, 0.f, 0.f, 0.f, 0.f, 0.f, 0.f, 0.f};
  for (int kt = 0; kt < Cpad; kt += 32) {
    // A 16x32 f16: lane row = l15, K chunks {kb..kb+7, 16+kb..16+kb+7}, kb = 8*hi
    v8h a0 = *(const v8h*)(xr + kt + hi * 8);
    v8h a1 = *(const v8h*)(xr + kt + 16 + hi * 8);
    v16h a = __builtin_shufflevector(a0, a1, 0,1,2,3,4,5,6,7,8,9,10,11,12,13,14,15);
    // B 32x16 f16: lane col = l15, K chunks {kb..kb+7, 16+kb..16+kb+7}, kb = 8*hi
    v8h b0 = *(const v8h*)(wr + kt + hi * 8);
    v8h b1 = *(const v8h*)(wr + kt + 16 + hi * 8);
    v16h bm = __builtin_shufflevector(b0, b1, 0,1,2,3,4,5,6,7,8,9,10,11,12,13,14,15);
    c = __builtin_amdgcn_wmma_f32_16x16x32_f16(false, a, false, bm, (short)0, c,
                                               false, false);
  }
  float bv = bias[ntile * 16 + l15];
  const int rbase = mtile * 16 + hi * 8;
#pragma unroll
  for (int r = 0; r < 8; ++r)
    Y[(size_t)(rbase + r) * Co + ntile * 16 + l15] = c[r] + bv;
}

// ---------------- per-channel sum / sumsq ----------------
__global__ void k_bn_stats(const float* __restrict__ Y, float* __restrict__ stats,
                           int M, int Co) {
  int t  = blockIdx.x * blockDim.x + threadIdx.x;   // launched <<<Co,256>>>
  int co = t % Co;
  int r0 = t / Co;
  int rs = (gridDim.x * blockDim.x) / Co;           // == 256
  float s = 0.f, s2 = 0.f;
  for (int r = r0; r < M; r += rs) {
    float v = Y[(size_t)r * Co + co];
    s += v; s2 += v * v;
  }
  atomicAdd(&stats[co], s);
  atomicAdd(&stats[Co + co], s2);
}

// ---------------- BN(batch stats) + ReLU; writes f32 in place + padded f16 next input ----------------
__global__ void k_bn_relu(float* __restrict__ Y, const float* __restrict__ stats,
                          const float* __restrict__ gamma, const float* __restrict__ beta,
                          int M, int Co, int CW, int writeNext,
                          _Float16* __restrict__ Xn, float invM) {
  size_t i = (size_t)blockIdx.x * blockDim.x + threadIdx.x;
  if (i >= (size_t)M * CW) return;
  int c = (int)(i % CW);
  size_t r = i / CW;
  if (c < Co) {
    float mu  = stats[c] * invM;
    float var = stats[Co + c] * invM - mu * mu;
    float v = (Y[r * Co + c] - mu) * rsqrtf(var + BN_EPS) * gamma[c] + beta[c];
    v = fmaxf(v, 0.f);
    Y[r * Co + c] = v;
    if (writeNext) Xn[r * CW + c] = (_Float16)v;
  } else {
    Xn[r * CW + c] = (_Float16)0.f;             // zero pad columns for next WMMA K-loop
  }
}

// ---------------- max over K neighbors ----------------
__global__ void k_maxpool(const float* __restrict__ Y, float* __restrict__ outP,
                          int BS, int K, int Co, int Ctot, int cbase) {
  int i = blockIdx.x * blockDim.x + threadIdx.x;
  if (i >= BS * Co) return;
  int c = i % Co, p = i / Co;
  float m = -1e30f;
  for (int k = 0; k < K; ++k)
    m = fmaxf(m, Y[((size_t)p * K + k) * Co + c]);
  outP[(size_t)p * Ctot + cbase + c] = m;
}

// =======================================================================
extern "C" void kernel_launch(void* const* d_in, const int* in_sizes, int n_in,
                              void* d_out, int out_size, void* d_ws, size_t ws_size,
                              hipStream_t stream) {
  (void)in_sizes; (void)n_in; (void)out_size; (void)ws_size;
  const float* xyz_in = (const float*)d_in[0];
  auto P = [&](int l, int w) { return (const float*)d_in[1 + l * 4 + w]; };

  char* ws = (char*)d_ws;
  size_t off = 0;
  auto alloc = [&](size_t n) { size_t o = off; off = (off + n + 255) & ~(size_t)255; return o; };
  float* xyz_t = (float*)(ws + alloc((size_t)8 * 8192 * 3 * 4));
  int*   fidx  = (int*)  (ws + alloc((size_t)8 * 1024 * 4));
  float* nx1   = (float*)(ws + alloc((size_t)8 * 1024 * 3 * 4));
  float* nx2   = (float*)(ws + alloc((size_t)8 * 512 * 3 * 4));
  float* l1p   = (float*)(ws + alloc((size_t)8 * 1024 * 96 * 4));
  float* l2p   = (float*)(ws + alloc((size_t)8 * 512 * 256 * 4));
  float* stats = (float*)(ws + alloc(2 * 128 * 4));
  _Float16* wt[12];
  for (int i = 0; i < 12; ++i) wt[i] = (_Float16*)(ws + alloc((size_t)128 * 128 * 2));
  _Float16* Xa = (_Float16*)(ws + alloc((size_t)33554432));
  _Float16* Xb = (_Float16*)(ws + alloc((size_t)33554432));
  float*    Yb = (float*)   (ws + alloc((size_t)67108864));

  struct LCfg { int Co, Cin, Cpad; };
  const LCfg Ls[12] = {
    {16, 6, 32}, {16, 16, 32}, {32, 16, 32},      // SA1 scale 1
    {32, 6, 32}, {32, 32, 32}, {64, 32, 32},      // SA1 scale 2
    {64, 99, 128}, {64, 64, 64}, {128, 64, 64},   // SA2 scale 1
    {64, 99, 128}, {96, 64, 64}, {128, 96, 96},   // SA2 scale 2
  };

  { size_t tot = (size_t)8 * 3 * 8192;
    k_t_in<<<(unsigned)((tot + 255) / 256), 256, 0, stream>>>(xyz_in, xyz_t, 8, 3, 8192); }

  k_fps<8><<<8, 1024, 0, stream>>>(xyz_t, 8192, 1024, fidx, nx1);

  auto run_scale = [&](const float* xyzP, const float* ptsP, const float* nxP,
                       int N, int S, int K, int D, int l0, float radius,
                       float* outP, int Ctot, int cbase) {
    int BS = 8 * S;
    int M  = BS * K;
    k_ball_group<<<BS / 8, 256, 0, stream>>>(xyzP, nxP, ptsP, N, S, K, D,
                                             Ls[l0].Cpad, radius * radius, Xa);
    _Float16* Xc = Xa;
    for (int l = 0; l < 3; ++l) {
      const LCfg& lay = Ls[l0 + l];
      k_pack_w<<<(lay.Co * lay.Cpad + 255) / 256, 256, 0, stream>>>(
          P(l0 + l, 0), wt[l0 + l], lay.Co, lay.Cin, lay.Cpad);
      k_zero<<<1, 256, 0, stream>>>(stats, 2 * lay.Co);
      dim3 g((M + 63) / 64, lay.Co / 16);
      k_gemm<<<g, 128, 0, stream>>>(Xc, wt[l0 + l], P(l0 + l, 1), Yb, M, lay.Cpad, lay.Co);
      k_bn_stats<<<lay.Co, 256, 0, stream>>>(Yb, stats, M, lay.Co);
      int wn = (l < 2) ? 1 : 0;
      int CW = wn ? Ls[l0 + l + 1].Cpad : lay.Co;
      _Float16* Xn = (Xc == Xa) ? Xb : Xa;
      size_t tot = (size_t)M * CW;
      k_bn_relu<<<(unsigned)((tot + 255) / 256), 256, 0, stream>>>(
          Yb, stats, P(l0 + l, 2), P(l0 + l, 3), M, lay.Co, CW, wn, Xn, 1.0f / (float)M);
      if (wn) Xc = Xn;
    }
    int lastCo = Ls[l0 + 2].Co;
    k_maxpool<<<(unsigned)(((size_t)BS * lastCo + 255) / 256), 256, 0, stream>>>(
        Yb, outP, BS, K, lastCo, Ctot, cbase);
  };

  // SA1 (N=8192 -> S=1024), points == xyz (D=3)
  run_scale(xyz_t, xyz_t, nx1, 8192, 1024, 16, 3, 0, 0.05f, l1p, 96, 0);
  run_scale(xyz_t, xyz_t, nx1, 8192, 1024, 32, 3, 3, 0.10f, l1p, 96, 32);

  // SA2 (N=1024 -> S=512), points = l1_pts (D=96)
  k_fps<1><<<8, 1024, 0, stream>>>(nx1, 1024, 512, fidx, nx2);
  run_scale(nx1, l1p, nx2, 1024, 512, 16, 96, 6, 0.10f, l2p, 256, 0);
  run_scale(nx1, l1p, nx2, 1024, 512, 32, 96, 9, 0.20f, l2p, 256, 128);

  // outputs: l2_xyz^T [8,3,512] then l2_pts^T [8,256,512]
  float* outF = (float*)d_out;
  { size_t tot = (size_t)8 * 512 * 3;
    k_t_out<<<(unsigned)((tot + 255) / 256), 256, 0, stream>>>(nx2, outF, 8, 512, 3); }
  { size_t tot = (size_t)8 * 512 * 256;
    k_t_out<<<(unsigned)((tot + 255) / 256), 256, 0, stream>>>(
        l2p, outF + (size_t)8 * 3 * 512, 8, 512, 256); }
}
